// NewAttentionBlock_50740743635121
// MI455X (gfx1250) — matrix operand order repaired and checked by reading, not compiled, not run on hardware
//
#include <hip/hip_runtime.h>
#include <hip/hip_bf16.h>

typedef __bf16 bf16_t;
typedef __attribute__((ext_vector_type(16))) __bf16 v16bf;
typedef __attribute__((ext_vector_type(8)))  float  v8f;
typedef __attribute__((ext_vector_type(4)))  unsigned int u32x4;
typedef __attribute__((ext_vector_type(8)))  int i32x8;
typedef __attribute__((ext_vector_type(4)))  int i32x4;

// ---------------------------------------------------------------- helpers ---
__device__ __forceinline__ bf16_t f2bf(float f) {
    union { float f; unsigned u; } a; a.f = f;
    unsigned u = a.u;
    u += 0x7FFFu + ((u >> 16) & 1u);          // round-to-nearest-even
    union { unsigned short s; bf16_t b; } r;
    r.s = (unsigned short)(u >> 16);
    return r.b;
}

__device__ __forceinline__ float block_reduce(float v, float* sm, bool do_max) {
    const int t = threadIdx.x;
    sm[t] = v;
    __syncthreads();
    for (int s = 128; s > 0; s >>= 1) {
        if (t < s) sm[t] = do_max ? fmaxf(sm[t], sm[t + s]) : (sm[t] + sm[t + s]);
        __syncthreads();
    }
    float r = sm[0];
    __syncthreads();
    return r;
}

// ------------------------------------------------------------ elementwise ---
__global__ __launch_bounds__(256) void zerof_kernel(float* p) { *p = 0.f; }

__global__ __launch_bounds__(256) void w2bf_kernel(const float* __restrict__ in,
                                                   bf16_t* __restrict__ out) {
    long long idx = (long long)blockIdx.x * 256 + threadIdx.x;
    out[idx] = f2bf(in[idx]);
}

__global__ __launch_bounds__(256) void prep_kernel(
    const float* __restrict__ x, const float* __restrict__ pe,
    const float* __restrict__ a1, const float* __restrict__ w1, const float* __restrict__ b1,
    const float* __restrict__ a3, const float* __restrict__ w3, const float* __restrict__ b3,
    float* __restrict__ prior, bf16_t* __restrict__ xn) {
    long long idx = (long long)blockIdx.x * 256 + threadIdx.x;
    int d = (int)(idx & 1023);                   // D = 1024
    float v = x[idx] + pe[d];
    prior[idx] = tanhf(a3[0] * v) * w3[d] + b3[d];
    xn[idx]    = f2bf(tanhf(a1[0] * v) * w1[d] + b1[d]);
}

__global__ __launch_bounds__(256) void dyt2_kernel(
    const float* __restrict__ in, const float* __restrict__ a,
    const float* __restrict__ w, const float* __restrict__ b,
    bf16_t* __restrict__ out) {
    long long idx = (long long)blockIdx.x * 256 + threadIdx.x;
    int d = (int)(idx & 1023);
    out[idx] = f2bf(tanhf(a[0] * in[idx]) * w[d] + b[d]);
}

// --------------------------------------------------------- WMMA GEMM core ---
constexpr int BM = 128, BN = 128, BK = 32, PAD = 8, LDR = BK + PAD;

enum { EPI_BF16 = 0, EPI_SCORES = 1, EPI_BIAS_F32 = 2,
       EPI_BIAS_GELU_BF16 = 3, EPI_BIAS_ADD_F32 = 4 };

// fragments per CDNA5 ISA (wave32):
//  A 16x32: lanes 0-15 -> K 0-7 & 16-23 ; lanes 16-31 -> K 8-15 & 24-31
//  B 32x16: lanes 0-15 -> K 0-7 & 16-23 ; lanes 16-31 -> K 8-15 & 24-31 (N = lane&15)
__device__ __forceinline__ void mma_tiles(const bf16_t As[][LDR], const bf16_t Bs[][LDR],
                                          v8f acc[2][4], int wm, int wn, int l16, int hs) {
    union Frag { v16bf v; float4 f4[2]; };
    Frag a[2], b[4];
#pragma unroll
    for (int mt = 0; mt < 2; mt++) {
        const bf16_t* p = &As[wm + mt * 16 + l16][hs * 8];
        a[mt].f4[0] = *(const float4*)p;
        a[mt].f4[1] = *(const float4*)(p + 16);
    }
#pragma unroll
    for (int nt = 0; nt < 4; nt++) {
        const bf16_t* p = &Bs[wn + nt * 16 + l16][hs * 8];
        b[nt].f4[0] = *(const float4*)p;
        b[nt].f4[1] = *(const float4*)(p + 16);
    }
#pragma unroll
    for (int mt = 0; mt < 2; mt++)
#pragma unroll
        for (int nt = 0; nt < 4; nt++)
            acc[mt][nt] = __builtin_amdgcn_wmma_f32_16x16x32_bf16(
                false, a[mt].v, false, b[nt].v, (short)0, acc[mt][nt], false, false);
}

template <int EPI>
__device__ __forceinline__ void epilogue(v8f acc[2][4], void* Cout,
    const float* bias, const float* add, int m0, int n0, int wm, int wn,
    int l16, int hs, int ldc, long long cbase, float scale) {
#pragma unroll
    for (int mt = 0; mt < 2; mt++) {
#pragma unroll
        for (int nt = 0; nt < 4; nt++) {
            const int gm = m0 + wm + mt * 16 + hs * 8;
            const int gn = n0 + wn + nt * 16 + l16;
#pragma unroll
            for (int r = 0; r < 8; r++) {
                const int row = gm + r;
                float c = acc[mt][nt][r];
                long long cidx = cbase + (long long)row * ldc + gn;
                if constexpr (EPI == EPI_BF16) {
                    ((bf16_t*)Cout)[cidx] = f2bf(c);
                } else if constexpr (EPI == EPI_SCORES) {
                    ((float*)Cout)[cidx] = c * scale - fabsf((float)(row - gn));
                } else if constexpr (EPI == EPI_BIAS_F32) {
                    ((float*)Cout)[cidx] = c + bias[gn];
                } else if constexpr (EPI == EPI_BIAS_GELU_BF16) {
                    float v = c + bias[gn];
                    float g = 0.5f * v * (1.0f + erff(v * 0.7071067811865475f));
                    ((bf16_t*)Cout)[cidx] = f2bf(g);
                } else {  // EPI_BIAS_ADD_F32 : + prior residual -> posterior
                    ((float*)Cout)[cidx] = c + bias[gn] + add[cidx];
                }
            }
        }
    }
}

// ------------------------------------------------ TDM tile load (D# pack) ---
// 2D tile: tile_dim0=BK elems along K, tile_dim1=BM rows, data_size=2B.
// LDS padding: every 16 DWORDs (one 32-elem bf16 row) insert 4 DWORDs -> LDR=40.
__device__ __forceinline__ void tdm_load_tile(const void* lds_dst,
                                              const bf16_t* gsrc, int ld) {
    unsigned long long ga = (unsigned long long)gsrc;
    unsigned lds = (unsigned)(unsigned long long)lds_dst;   // low32 = LDS offset
    u32x4 g0;
    g0[0] = 1u;                                             // count=1, user mode
    g0[1] = lds;                                            // lds_addr
    g0[2] = (unsigned)(ga & 0xFFFFFFFFu);                   // global_addr lo
    g0[3] = (unsigned)((ga >> 32) & 0x01FFFFFFu) | (2u << 30);  // addr hi | type=2
    const unsigned d0 = 1u << 30, d1 = 1u << 30;            // tensor dims (no OOB)
    i32x8 g1;
    g1[0] = (int)((1u << 16)        // data_size = 2 bytes
                | (1u << 20)        // pad_enable
                | (3u << 22)        // pad_interval: 16 DWORDs
                | (3u << 25));      // pad_amount: 4 DWORDs
    g1[1] = (int)((d0 & 0xFFFFu) << 16);
    g1[2] = (int)((d0 >> 16) | ((d1 & 0xFFFFu) << 16));
    g1[3] = (int)((d1 >> 16) | ((unsigned)BK << 16));       // tile_dim0 = 32
    g1[4] = (int)(unsigned)BM;                              // tile_dim1 = 128
    g1[5] = (int)(unsigned)ld;                              // dim0 stride lo
    g1[6] = 0;                                              // stride hi / dim1 stride lo
    g1[7] = 0;
    i32x4 z4 = {0, 0, 0, 0};
#if __clang_major__ >= 23
    i32x8 z8 = {0, 0, 0, 0, 0, 0, 0, 0};
    __builtin_amdgcn_tensor_load_to_lds(g0, g1, z4, z4, z8, 0);
#else
    __builtin_amdgcn_tensor_load_to_lds(g0, g1, z4, z4, 0);
#endif
}

// ------------------------------- GEMM, TDM-staged (B in [N,K] weight form) ---
template <int EPI>
__global__ __launch_bounds__(256) void gemm_wmma_tdm(
    const bf16_t* __restrict__ A, const bf16_t* __restrict__ Bw,
    const float* __restrict__ bias, const float* __restrict__ add,
    void* __restrict__ Cout,
    int K, int lda, int ldb, int ldc,
    long long aBatch, long long bBatch, long long cBatch, float scale) {
    __shared__ bf16_t As[2][BM][LDR];
    __shared__ bf16_t Bs[2][BN][LDR];

    const int tid  = threadIdx.x;
    const int lane = tid & 31, l16 = lane & 15, hs = lane >> 4;
    const int wave = tid >> 5;
    const int wm = (wave & 3) * 32, wn = (wave >> 2) * 64;
    const int m0 = blockIdx.y * BM, n0 = blockIdx.x * BN;
    const int bz = blockIdx.z;
    A  += (long long)bz * aBatch;
    Bw += (long long)bz * bBatch;
    const bf16_t* Arow = A  + (long long)m0 * lda;
    const bf16_t* Brow = Bw + (long long)n0 * ldb;

    v8f acc[2][4];
#pragma unroll
    for (int i = 0; i < 2; i++)
#pragma unroll
        for (int j = 0; j < 4; j++)
            acc[i][j] = (v8f){0.f, 0.f, 0.f, 0.f, 0.f, 0.f, 0.f, 0.f};

    const int nkt = K / BK;
    if (tid < 32) {                       // wave 0 drives the Tensor Data Mover
        tdm_load_tile(&As[0][0][0], Arow, lda);
        tdm_load_tile(&Bs[0][0][0], Brow, ldb);
    }
    for (int kt = 0; kt < nkt; ++kt) {
        const int cur = kt & 1, nxt = cur ^ 1;
        if (tid < 32) {
            if (kt + 1 < nkt) {           // prefetch next pair into other buffer
                tdm_load_tile(&As[nxt][0][0], Arow + (kt + 1) * BK, lda);
                tdm_load_tile(&Bs[nxt][0][0], Brow + (kt + 1) * BK, ldb);
                __builtin_amdgcn_s_wait_tensorcnt(2);  // in-order: current pair done
            } else {
                __builtin_amdgcn_s_wait_tensorcnt(0);
            }
        }
        __syncthreads();
        mma_tiles(As[cur], Bs[cur], acc, wm, wn, l16, hs);
        __syncthreads();                  // buffer free before TDM overwrites it
    }
    epilogue<EPI>(acc, Cout, bias, add, m0, n0, wm, wn, l16, hs, ldc,
                  (long long)bz * cBatch, scale);
}

// --------------------- GEMM, manual staging (B in [K,N] form, needs transp) --
template <int EPI>
__global__ __launch_bounds__(256) void gemm_wmma_kn(
    const bf16_t* __restrict__ A, const bf16_t* __restrict__ Bw,
    const float* __restrict__ bias, const float* __restrict__ add,
    void* __restrict__ Cout,
    int K, int lda, int ldb, int ldc,
    long long aBatch, long long bBatch, long long cBatch, float scale) {
    __shared__ bf16_t As[BM][LDR];
    __shared__ bf16_t Bs[BN][LDR];

    const int tid  = threadIdx.x;
    const int lane = tid & 31, l16 = lane & 15, hs = lane >> 4;
    const int wave = tid >> 5;
    const int wm = (wave & 3) * 32, wn = (wave >> 2) * 64;
    const int m0 = blockIdx.y * BM, n0 = blockIdx.x * BN;
    const int bz = blockIdx.z;
    A  += (long long)bz * aBatch;
    Bw += (long long)bz * bBatch;

    v8f acc[2][4];
#pragma unroll
    for (int i = 0; i < 2; i++)
#pragma unroll
        for (int j = 0; j < 4; j++)
            acc[i][j] = (v8f){0.f, 0.f, 0.f, 0.f, 0.f, 0.f, 0.f, 0.f};

    const int nkt = K / BK;
    for (int kt = 0; kt < nkt; ++kt) {
        const int k0 = kt * BK;
#pragma unroll
        for (int i = 0; i < 2; i++) {            // A tile, 16B vector loads
            int c   = tid + i * 256;
            int row = c >> 2;
            int kc  = (c & 3) << 3;
            *(float4*)(&As[row][kc]) =
                *(const float4*)(A + (long long)(m0 + row) * lda + (k0 + kc));
        }
#pragma unroll
        for (int i = 0; i < 2; i++) {            // B [K,N] -> Bs[n][k]
            int c  = tid + i * 256;
            int kr = c >> 4;
            int nc = (c & 15) << 3;
            union { float4 f; bf16_t e[8]; } u;
            u.f = *(const float4*)(Bw + (long long)(k0 + kr) * ldb + (n0 + nc));
#pragma unroll
            for (int j = 0; j < 8; j++) Bs[nc + j][kr] = u.e[j];
        }
        if (kt + 1 < nkt)
            __builtin_prefetch(A + (long long)(m0 + (tid >> 2)) * lda + (k0 + BK), 0, 1);
        __syncthreads();
        mma_tiles(As, Bs, acc, wm, wn, l16, hs);
        __syncthreads();
    }
    epilogue<EPI>(acc, Cout, bias, add, m0, n0, wm, wn, l16, hs, ldc,
                  (long long)bz * cBatch, scale);
}

// ------------------------------------------------ softmax (in-place P<-S) ---
__global__ __launch_bounds__(256) void softmax_rows(float* __restrict__ S) {
    __shared__ float sm[256];
    const int t = threadIdx.x;
    const long long rb = (long long)blockIdx.x * 2048;
    float* row = S + rb;
    float v[8];
    float mx = -3.4e38f;
#pragma unroll
    for (int i = 0; i < 8; i++) { v[i] = row[t + i * 256]; mx = fmaxf(mx, v[i]); }
    mx = block_reduce(mx, sm, true);        // all reads complete at this barrier
    float s = 0.f;
#pragma unroll
    for (int i = 0; i < 8; i++) { v[i] = expf(v[i] - mx); s += v[i]; }
    s = block_reduce(s, sm, false);
    const float inv = 1.0f / s;
    bf16_t* prow = ((bf16_t*)S) + rb * 2;   // bf16 alias, row stride 2*T elems
#pragma unroll
    for (int i = 0; i < 8; i++) prow[t + i * 256] = f2bf(v[i] * inv);
}

// --------------------------------------------------------------- JS loss ----
__global__ __launch_bounds__(256) void js_loss_kernel(
    const float* __restrict__ prior, const float* __restrict__ post,
    float* __restrict__ loss, float coef) {
    __shared__ float sm[256];
    const int t = threadIdx.x;
    const long long rb = (long long)blockIdx.x * 1024;
    float pr[4], po[4], m[4];
    float mx_pr = -3.4e38f, mx_po = -3.4e38f, mx_m = -3.4e38f;
#pragma unroll
    for (int i = 0; i < 4; i++) {
        pr[i] = prior[rb + t + i * 256];
        po[i] = post[rb + t + i * 256];
        m[i]  = 0.5f * (pr[i] + po[i]);
        mx_pr = fmaxf(mx_pr, pr[i]);
        mx_po = fmaxf(mx_po, po[i]);
        mx_m  = fmaxf(mx_m, m[i]);
    }
    mx_pr = block_reduce(mx_pr, sm, true);
    mx_po = block_reduce(mx_po, sm, true);
    mx_m  = block_reduce(mx_m, sm, true);
    float s_pr = 0.f, s_po = 0.f, s_m = 0.f;
#pragma unroll
    for (int i = 0; i < 4; i++) {
        s_pr += expf(pr[i] - mx_pr);
        s_po += expf(po[i] - mx_po);
        s_m  += expf(m[i]  - mx_m);
    }
    s_pr = block_reduce(s_pr, sm, false);
    s_po = block_reduce(s_po, sm, false);
    s_m  = block_reduce(s_m,  sm, false);
    const float lse_pr = mx_pr + logf(s_pr);
    const float lse_po = mx_po + logf(s_po);
    const float lse_m  = mx_m  + logf(s_m);
    float acc = 0.f;
#pragma unroll
    for (int i = 0; i < 4; i++) {
        float lm = m[i] - lse_m;
        float pm = expf(lm);
        acc += pm * (2.f * lm - (pr[i] - lse_pr) - (po[i] - lse_po));
    }
    acc = block_reduce(acc, sm, false);
    if (t == 0) atomicAdd(loss, acc * coef);
}

// ----------------------------------------------------------------- launch ---
extern "C" void kernel_launch(void* const* d_in, const int* in_sizes, int n_in,
                              void* d_out, int out_size, void* d_ws, size_t ws_size,
                              hipStream_t stream) {
    (void)in_sizes; (void)n_in; (void)out_size; (void)ws_size;
    const float* x  = (const float*)d_in[0];
    const float* pe = (const float*)d_in[1];
    const float* a1 = (const float*)d_in[2];
    const float* w1 = (const float*)d_in[3];
    const float* b1 = (const float*)d_in[4];
    const float* a2 = (const float*)d_in[5];
    const float* w2 = (const float*)d_in[6];
    const float* b2 = (const float*)d_in[7];
    const float* a3 = (const float*)d_in[8];
    const float* w3 = (const float*)d_in[9];
    const float* b3 = (const float*)d_in[10];
    const float* Wq = (const float*)d_in[11];
    const float* Wk = (const float*)d_in[12];
    const float* Wv = (const float*)d_in[13];
    const float* Wp = (const float*)d_in[14];
    const float* bp = (const float*)d_in[15];
    const float* We = (const float*)d_in[16];
    const float* be = (const float*)d_in[17];
    const float* Wc = (const float*)d_in[18];
    const float* bc = (const float*)d_in[19];

    constexpr long long Bz = 4, T = 2048, D = 1024, DFF = 4096;
    constexpr long long BT = Bz * T, BTD = BT * D;

    char* ws = (char*)d_ws;
    auto alloc = [&](size_t bytes) -> char* {
        char* p = ws;
        ws += (bytes + 255) & ~(size_t)255;
        return p;
    };
    float*  prior = (float*) alloc(BTD * 4);
    bf16_t* xn    = (bf16_t*)alloc(BTD * 2);
    bf16_t* wqb   = (bf16_t*)alloc(D * D * 2);
    bf16_t* wkb   = (bf16_t*)alloc(D * D * 2);
    bf16_t* wvb   = (bf16_t*)alloc(D * D * 2);
    bf16_t* wpb   = (bf16_t*)alloc(D * D * 2);
    bf16_t* web   = (bf16_t*)alloc(DFF * D * 2);
    bf16_t* wcb   = (bf16_t*)alloc(D * DFF * 2);
    bf16_t* Qb    = (bf16_t*)alloc(BTD * 2);
    bf16_t* Kb    = (bf16_t*)alloc(BTD * 2);
    bf16_t* Vb    = (bf16_t*)alloc(BTD * 2);
    float*  S     = (float*) alloc(Bz * T * T * 4);   // scores
    bf16_t* P     = (bf16_t*)S;                       // softmax result (in-place alias)
    bf16_t* ff    = (bf16_t*)S;                       // expand output reuses S (same 64MB)
    bf16_t* attn  = (bf16_t*)alloc(BTD * 2);
    float*  projo = (float*) alloc(BTD * 4);
    bf16_t* h     = (bf16_t*)alloc(BTD * 2);

    float* post = (float*)d_out;
    float* loss = post + BTD;

    zerof_kernel<<<1, 256, 0, stream>>>(loss);
    prep_kernel<<<(int)(BTD / 256), 256, 0, stream>>>(x, pe, a1, w1, b1, a3, w3, b3,
                                                      prior, xn);
    w2bf_kernel<<<(int)(D * D / 256),   256, 0, stream>>>(Wq, wqb);
    w2bf_kernel<<<(int)(D * D / 256),   256, 0, stream>>>(Wk, wkb);
    w2bf_kernel<<<(int)(D * D / 256),   256, 0, stream>>>(Wv, wvb);
    w2bf_kernel<<<(int)(D * D / 256),   256, 0, stream>>>(Wp, wpb);
    w2bf_kernel<<<(int)(DFF * D / 256), 256, 0, stream>>>(We, web);
    w2bf_kernel<<<(int)(D * DFF / 256), 256, 0, stream>>>(Wc, wcb);

    // Q/K/V = xn @ W^T  (M=8192, N=1024, K=1024) — TDM-staged
    {
        dim3 g((int)(D / 128), (int)(BT / 128), 1);
        gemm_wmma_tdm<EPI_BF16><<<g, 256, 0, stream>>>(
            xn, wqb, nullptr, nullptr, Qb, (int)D, (int)D, (int)D, (int)D,
            0, 0, 0, 1.f);
        gemm_wmma_tdm<EPI_BF16><<<g, 256, 0, stream>>>(
            xn, wkb, nullptr, nullptr, Kb, (int)D, (int)D, (int)D, (int)D,
            0, 0, 0, 1.f);
        gemm_wmma_tdm<EPI_BF16><<<g, 256, 0, stream>>>(
            xn, wvb, nullptr, nullptr, Vb, (int)D, (int)D, (int)D, (int)D,
            0, 0, 0, 1.f);
    }
    // scores: S[b,t,s] = (Q.K^T)/32 - |t-s|   (batched)
    {
        dim3 g((int)(T / 128), (int)(T / 128), (int)Bz);
        gemm_wmma_tdm<EPI_SCORES><<<g, 256, 0, stream>>>(
            Qb, Kb, nullptr, nullptr, S, (int)D, (int)D, (int)D, (int)T,
            T * D, T * D, T * T, 0.03125f);
    }
    softmax_rows<<<(int)(Bz * T), 256, 0, stream>>>(S);
    // attn_out = P @ V  (A lda = 2T bf16 over f32 buffer; B is [K,N] row-major)
    {
        dim3 g((int)(D / 128), (int)(T / 128), (int)Bz);
        gemm_wmma_kn<EPI_BF16><<<g, 256, 0, stream>>>(
            P, Vb, nullptr, nullptr, attn, (int)T, (int)(2 * T), (int)D, (int)D,
            2 * T * T, T * D, T * D, 1.f);
    }
    // proj: out = attn @ Wp^T + bp (f32)
    {
        dim3 g((int)(D / 128), (int)(BT / 128), 1);
        gemm_wmma_tdm<EPI_BIAS_F32><<<g, 256, 0, stream>>>(
            attn, wpb, bp, nullptr, projo, (int)D, (int)D, (int)D, (int)D,
            0, 0, 0, 1.f);
    }
    dyt2_kernel<<<(int)(BTD / 256), 256, 0, stream>>>(projo, a2, w2, b2, h);
    // expand + exact GELU (bf16 out, reuses S region)
    {
        dim3 g((int)(DFF / 128), (int)(BT / 128), 1);
        gemm_wmma_tdm<EPI_BIAS_GELU_BF16><<<g, 256, 0, stream>>>(
            h, web, be, nullptr, ff, (int)D, (int)D, (int)D, (int)DFF,
            0, 0, 0, 1.f);
    }
    // contract + bias + prior residual -> posterior (d_out)
    {
        dim3 g((int)(D / 128), (int)(BT / 128), 1);
        gemm_wmma_tdm<EPI_BIAS_ADD_F32><<<g, 256, 0, stream>>>(
            ff, wcb, bc, prior, post, (int)DFF, (int)DFF, (int)DFF, (int)D,
            0, 0, 0, 1.f);
    }
    js_loss_kernel<<<(int)BT, 256, 0, stream>>>(prior, post, loss, 0.5f / (float)Bz);
}
